// PrimaryNetwork_38766374813849
// MI455X (gfx1250) — matrix operand order repaired and checked by reading, not compiled, not run on hardware
//
#include <hip/hip_runtime.h>

// ---------------------------------------------------------------------------
// Types / helpers
// ---------------------------------------------------------------------------
typedef __attribute__((ext_vector_type(16))) __bf16        bf16x16;
typedef __attribute__((ext_vector_type(8)))  float         f32x8;
typedef __attribute__((ext_vector_type(8)))  unsigned int  u32x8;

__device__ __forceinline__ unsigned short f2bf(float f) {
  unsigned u = __builtin_bit_cast(unsigned, f);
  u += 0x7FFFu + ((u >> 16) & 1u);          // round-to-nearest-even
  return (unsigned short)(u >> 16);
}
__device__ __forceinline__ float bf2f(unsigned short h) {
  unsigned u = ((unsigned)h) << 16;
  return __builtin_bit_cast(float, u);
}

// ---------------------------------------------------------------------------
// Hypernetwork (tiny GEMMs, scalar is fine)
// ---------------------------------------------------------------------------
__global__ void hyp_c_kernel(const float* __restrict__ ym, const float* __restrict__ key,
                             const float* __restrict__ Wimg, const float* __restrict__ bimg,
                             const float* __restrict__ Wkey, const float* __restrict__ bkey,
                             float* __restrict__ cvec) {
  int d = threadIdx.x;                       // 64 threads
  float acc = bimg[d] + bkey[d];
  for (int i = 0; i < 3888; ++i) acc += ym[i] * Wimg[i * 64 + d];
  for (int t = 0; t < 64;   ++t) acc += key[t] * Wkey[t * 64 + d];
  cvec[d] = acc;
}

__global__ void hyp_h_kernel(const float* __restrict__ z_all, const float* __restrict__ cvec,
                             const float* __restrict__ W2, const float* __restrict__ B2,
                             float* __restrict__ h) {
  int tid = blockIdx.x * 256 + threadIdx.x;
  if (tid >= 242 * 1024) return;
  int z = tid >> 10, u = tid & 1023;
  float acc = B2[u];
  for (int t = 0; t < 64; ++t) acc += (z_all[z * 64 + t] + cvec[t]) * W2[t * 1024 + u];
  h[tid] = acc;
}

__global__ void hyp_kflat_kernel(const float* __restrict__ h, const float* __restrict__ W1,
                                 const float* __restrict__ B1, float* __restrict__ kflat) {
  int tid = blockIdx.x * 256 + threadIdx.x;
  if (tid >= 3872 * 144) return;
  int r = tid / 144, v = tid % 144;
  const float* hr = h + (r >> 4) * 1024 + (r & 15) * 64;
  float acc = B1[v];
  for (int t = 0; t < 64; ++t) acc += hr[t] * W1[t * 144 + v];
  kflat[tid] = acc;
}

// ---------------------------------------------------------------------------
// Weight packing into WMMA A-matrix per-lane layout (bf16)
// pack[((mt*KT+kt)*32 + lane)*16 + j] , A-layout: m=lane%16, g=lane/16,
// k = kt*32 + 16*g + j ; K ordering: k = (ky*3+kx)*CIN + ci
// ---------------------------------------------------------------------------
__global__ void pack_layer_kernel(const float* __restrict__ kflat,
                                  unsigned short* __restrict__ pack,
                                  int off_z, int hh, int kk) {
  const int CIN = kk * 16;
  const int K   = CIN * 9;
  const int KT  = (K + 31) / 32;
  const int total = hh * KT * 512;
  int tid = blockIdx.x * 256 + threadIdx.x;
  if (tid >= total) return;
  int j    = tid & 15;
  int lane = (tid >> 4) & 31;
  int tile = tid >> 9;
  int kt = tile % KT;
  int mt = tile / KT;
  int m = lane & 15, g = lane >> 4;
  int k = kt * 32 + (g << 4) + j;
  float val = 0.f;
  if (k < K) {
    int cig = k % CIN, p = k / CIN;       // p = ky*3+kx
    int b = cig >> 4, ci = cig & 15;
    int row = (off_z + mt * kk + b) * 16 + m;
    val = kflat[row * 144 + ci * 9 + p];
  }
  pack[tile * 512 + lane * 16 + j] = f2bf(val);
}

__global__ void pack_conv1_kernel(const float* __restrict__ w, unsigned short* __restrict__ pack) {
  int tid = threadIdx.x;                    // 512 threads
  int j = tid & 15, lane = tid >> 4;
  int m = lane & 15, g = lane >> 4;
  int k = (g << 4) + j;
  float val = 0.f;
  if (k < 27) {
    int ci = k % 3, p = k / 3;
    val = w[m * 27 + ci * 9 + p];          // conv1_w[co][ci][ky][kx]
  }
  pack[lane * 16 + j] = f2bf(val);
}

// ---------------------------------------------------------------------------
// Implicit-GEMM conv with v_wmma_f32_16x16x32_bf16  (NHWC activations)
// act: bf16 [512][H][W][CIN]; out: f32 [512][HOUT][WOUT][COUT]
// B operand per lane: ONE 32B vector load (16 contiguous channels at one tap)
// D store per tile: ONE contiguous f32x8 store (8 consecutive channels)
// ---------------------------------------------------------------------------
template <int CIN, int COUT, int H, int W, int STRIDE, int HOUT, int WOUT>
__global__ __launch_bounds__(256) void conv_wmma_kernel(
    const unsigned short* __restrict__ act,
    const unsigned short* __restrict__ wpack,
    float* __restrict__ out) {
  constexpr int K   = CIN * 9;
  constexpr int KT  = (K + 31) / 32;
  constexpr int MT  = COUT / 16;
  constexpr int HWO = HOUT * WOUT;

  const int wid  = threadIdx.x >> 5;
  const int lane = threadIdx.x & 31;
  const int g    = lane >> 4;
  const int n16  = lane & 15;
  const int col  = (blockIdx.x * 8 + wid) * 16 + n16;
  const int n    = col / HWO;
  const int rem  = col % HWO;
  const int y    = rem / WOUT;
  const int x    = rem % WOUT;

  f32x8 zero8;
#pragma unroll
  for (int t = 0; t < 8; ++t) zero8[t] = 0.f;
  f32x8 acc[MT];
#pragma unroll
  for (int m = 0; m < MT; ++m) acc[m] = zero8;

  for (int kt = 0; kt < KT; ++kt) {
    // A tiles: pre-swizzled, contiguous 32B per lane
    bf16x16 a[MT];
#pragma unroll
    for (int m = 0; m < MT; ++m) {
      const u32x8 av = *(const u32x8*)(wpack + (((m * KT + kt) * 32 + lane) << 4));
      a[m] = __builtin_bit_cast(bf16x16, av);
    }
    // B tile: lane = column, j -> K = kt*32 + 16*g + j
    bf16x16 bv;
    const int ks = kt * 32 + g * 16;
    if constexpr (CIN >= 16) {
      // 16-run of k = one filter tap p, 16 contiguous channels (NHWC)
      u32x8 raw;
#pragma unroll
      for (int t = 0; t < 8; ++t) raw[t] = 0u;
      if (ks < K) {
        const int p   = ks / CIN;
        const int ci0 = ks % CIN;
        const int ky = p / 3, kx = p % 3;
        const int yy = y * STRIDE + ky - 1;
        const int xx = x * STRIDE + kx - 1;
        if ((unsigned)yy < (unsigned)H && (unsigned)xx < (unsigned)W) {
          const long long idx = (((long long)n * H + yy) * W + xx) * CIN + ci0;
          raw = *(const u32x8*)(act + idx);         // 32B aligned (ci0 % 16 == 0)
        }
      }
      bv = __builtin_bit_cast(bf16x16, raw);
    } else {                                // conv1: CIN=3, K=27 (padded to 32)
      union { unsigned short u[16]; bf16x16 v; } bb;
#pragma unroll
      for (int j = 0; j < 16; ++j) {
        const int k = ks + j;
        unsigned short v = 0;
        if (k < K) {
          const int p = k / CIN, ci = k % CIN;
          const int ky = p / 3, kx = p % 3;
          const int yy = y * STRIDE + ky - 1;
          const int xx = x * STRIDE + kx - 1;
          if ((unsigned)yy < (unsigned)H && (unsigned)xx < (unsigned)W)
            v = act[(((long long)n * H + yy) * W + xx) * CIN + ci];
        }
        bb.u[j] = v;
      }
      bv = bb.v;
    }
#pragma unroll
    for (int m = 0; m < MT; ++m)
      acc[m] = __builtin_amdgcn_wmma_f32_16x16x32_bf16(
          false, a[m], false, bv, (short)0, acc[m], false, false);
  }
  // D layout: d[j] = row (m*16 + 8*g + j) = output channel -> contiguous NHWC store
  const long long obase = (((long long)n * HOUT + y) * WOUT + x) * COUT + 8 * g;
#pragma unroll
  for (int m = 0; m < MT; ++m)
    *(f32x8*)(out + obase + m * 16) = acc[m];
}

// ---------------------------------------------------------------------------
// BatchNorm over NHWC: deterministic reduction + fused apply/ReLU/residual
// C in {16,32,64} divides 256 -> each thread owns exactly one channel.
// ---------------------------------------------------------------------------
#define BN_SPLIT 64

__global__ void bn_reduce_kernel(const float* __restrict__ conv, float* __restrict__ part,
                                 int C, long long total) {
  const int s = blockIdx.x;                        // BN_SPLIT slices
  const long long chunk = total / BN_SPLIT;        // multiple of C
  const long long i0 = (long long)s * chunk;
  const long long i1 = i0 + chunk;
  float sum = 0.f, sq = 0.f;
  for (long long i = i0 + threadIdx.x; i < i1; i += 256) {   // coalesced
    float v = conv[i];
    sum += v; sq += v * v;
  }
  __shared__ float s1[256], s2[256];
  s1[threadIdx.x] = sum; s2[threadIdx.x] = sq;
  __syncthreads();
  for (int st = 128; st >= C; st >>= 1) {          // same-channel partners
    if (threadIdx.x < st) { s1[threadIdx.x] += s1[threadIdx.x + st];
                            s2[threadIdx.x] += s2[threadIdx.x + st]; }
    __syncthreads();
  }
  if (threadIdx.x < C) {
    part[(threadIdx.x * BN_SPLIT + s) * 2 + 0] = s1[threadIdx.x];
    part[(threadIdx.x * BN_SPLIT + s) * 2 + 1] = s2[threadIdx.x];
  }
}

__global__ void bn_finalize_kernel(const float* __restrict__ part,
                                   const float* __restrict__ gamma, const float* __restrict__ beta,
                                   float* __restrict__ scale, float* __restrict__ shift,
                                   int C, int HW) {
  int c = threadIdx.x;
  if (c >= C) return;
  float s = 0.f, q = 0.f;
  for (int i = 0; i < BN_SPLIT; ++i) {
    s += part[(c * BN_SPLIT + i) * 2 + 0];
    q += part[(c * BN_SPLIT + i) * 2 + 1];
  }
  float inv = 1.f / (512.f * (float)HW);
  float m = s * inv;
  float var = q * inv - m * m;
  float sc = gamma[c] * rsqrtf(var + 1e-5f);
  scale[c] = sc;
  shift[c] = beta[c] - m * sc;
}

__global__ void bn_apply_kernel(const float* __restrict__ conv,
                                const unsigned short* __restrict__ resid,
                                const float* __restrict__ scale, const float* __restrict__ shift,
                                unsigned short* __restrict__ outact,
                                long long total, int cmask) {
  for (long long i = (long long)blockIdx.x * 256 + threadIdx.x; i < total;
       i += (long long)gridDim.x * 256) {
    int c = (int)(i & cmask);                      // NHWC, C power of two
    float v = conv[i] * scale[c] + shift[c];
    if (resid) v += bf2f(resid[i]);
    v = fmaxf(v, 0.f);
    outact[i] = f2bf(v);
  }
}

// ---------------------------------------------------------------------------
// 1x1 stride-2 residual conv (NHWC), x convert NCHW->NHWC bf16, head
// ---------------------------------------------------------------------------
__global__ void res_conv_kernel(const unsigned short* __restrict__ act,
                                const float* __restrict__ w, const float* __restrict__ b,
                                unsigned short* __restrict__ out,
                                int CI, int H, int W, int CO, int HO, int WO, long long total) {
  long long i = (long long)blockIdx.x * 256 + threadIdx.x;
  if (i >= total) return;
  int co = (int)(i % CO); long long t = i / CO;
  int xo = (int)(t % WO); t /= WO;
  int yo = (int)(t % HO); int n = (int)(t / HO);
  float acc = b[co];
  const unsigned short* src = act + (((long long)n * H + yo * 2) * W + xo * 2) * CI;
  for (int ci = 0; ci < CI; ++ci) acc += bf2f(src[ci]) * w[co * CI + ci];
  out[i] = f2bf(acc);
}

__global__ void cvt_x_kernel(const float* __restrict__ in, unsigned short* __restrict__ out,
                             long long nelem) {
  // NCHW f32 [512][3][36][36] -> NHWC bf16 [512][36][36][3]
  long long i = (long long)blockIdx.x * 256 + threadIdx.x;
  if (i >= nelem) return;
  int c = (int)(i % 3); long long t = i / 3;
  int xx = (int)(t % 36); t /= 36;
  int yy = (int)(t % 36); int n = (int)(t / 36);
  out[i] = f2bf(in[(((long long)n * 3 + c) * 36 + yy) * 36 + xx]);
}

__global__ void final_head_kernel(const unsigned short* __restrict__ act,
                                  const float* __restrict__ Wf, const float* __restrict__ fb,
                                  float* __restrict__ out) {
  __shared__ float sm[64];
  int n = blockIdx.x;
  int c = threadIdx.x;                       // 64 threads
  float s = 0.f;
  const unsigned short* p = act + (long long)n * 81 * 64 + c;   // NHWC [512][9][9][64]
  for (int k = 0; k < 81; ++k) s += bf2f(p[k * 64]);            // coalesced over c
  sm[c] = s * (1.f / 81.f);
  __syncthreads();
  if (c < 20) {
    float acc = fb[c];
    for (int t = 0; t < 64; ++t) acc += sm[t] * Wf[c * 64 + t];
    out[n * 20 + c] = acc;
  }
}

// ---------------------------------------------------------------------------
// Host orchestration
// ---------------------------------------------------------------------------
extern "C" void kernel_launch(void* const* d_in, const int* in_sizes, int n_in,
                              void* d_out, int out_size, void* d_ws, size_t ws_size,
                              hipStream_t stream) {
  (void)in_sizes; (void)out_size; (void)ws_size;

  const float* x        = (const float*)d_in[0];
  const float* ym       = (const float*)d_in[1];
  const float* keyp     = (const float*)d_in[2];
  const float* z_all    = (const float*)d_in[3];
  const float* hWimg    = (const float*)d_in[4];
  const float* hbimg    = (const float*)d_in[5];
  const float* hWkey    = (const float*)d_in[6];
  const float* hbkey    = (const float*)d_in[7];
  const float* hW2      = (const float*)d_in[8];
  const float* hB2      = (const float*)d_in[9];
  const float* hW1      = (const float*)d_in[10];
  const float* hB1      = (const float*)d_in[11];
  const float* conv1w   = (const float*)d_in[12];
  // conv1_b (d_in[13]) is absorbed by the following batchnorm -> skipped.

  int BN_CH[37]; BN_CH[0] = 16;
  for (int i = 0; i < 18; ++i) {
    int ch = (i < 6) ? 16 : ((i < 12) ? 32 : 64);
    BN_CH[1 + 2 * i] = ch; BN_CH[2 + 2 * i] = ch;
  }
  const float *gam[37], *bet[37];
  int rbase;
  if (n_in >= 94) {                           // tuples expanded as leaves
    for (int j = 0; j < 37; ++j) { gam[j] = (const float*)d_in[14 + j];
                                   bet[j] = (const float*)d_in[51 + j]; }
    rbase = 88;
  } else {                                    // tuples concatenated fallback
    const float* G = (const float*)d_in[14];
    const float* B = (const float*)d_in[15];
    int off = 0;
    for (int j = 0; j < 37; ++j) { gam[j] = G + off; bet[j] = B + off; off += BN_CH[j]; }
    rbase = 16;
  }
  const float* res1w = (const float*)d_in[rbase + 0];
  const float* res1b = (const float*)d_in[rbase + 1];
  const float* res2w = (const float*)d_in[rbase + 2];
  const float* res2b = (const float*)d_in[rbase + 3];
  const float* finW  = (const float*)d_in[rbase + 4];
  const float* finB  = (const float*)d_in[rbase + 5];

  // ---- workspace carve --------------------------------------------------
  char* wsb = (char*)d_ws;
  size_t woff = 0;
  auto carve = [&](size_t bytes) -> void* {
    void* p = wsb + woff;
    woff += (bytes + 255) & ~(size_t)255;
    return p;
  };
  unsigned short* xbf   = (unsigned short*)carve(1990656ULL * 2);
  unsigned short* act0  = (unsigned short*)carve(10616832ULL * 2);  // h (NHWC bf16)
  unsigned short* act1  = (unsigned short*)carve(10616832ULL * 2);  // mid
  float*          convb = (float*)carve(10616832ULL * 4);           // raw conv f32
  unsigned short* resb  = (unsigned short*)carve(5308416ULL * 2);
  float*          cvec  = (float*)carve(64 * 4);
  float*          hbuf  = (float*)carve(242ULL * 1024 * 4);
  float*          kflat = (float*)carve(3872ULL * 144 * 4);
  float*          part  = (float*)carve(64ULL * BN_SPLIT * 2 * 4);
  float*          bnsc  = (float*)carve(64 * 4);
  float*          bnsh  = (float*)carve(64 * 4);
  unsigned short* pack  = (unsigned short*)carve(600000ULL * 2);

  // ---- hypernetwork -----------------------------------------------------
  hyp_c_kernel<<<1, 64, 0, stream>>>(ym, keyp, hWimg, hbimg, hWkey, hbkey, cvec);
  hyp_h_kernel<<<(242 * 1024 + 255) / 256, 256, 0, stream>>>(z_all, cvec, hW2, hB2, hbuf);
  hyp_kflat_kernel<<<(3872 * 144 + 255) / 256, 256, 0, stream>>>(hbuf, hW1, hB1, kflat);

  // ---- weight packing ---------------------------------------------------
  int HH[36], KK[36];
  for (int l = 0; l < 36; ++l) {
    if (l < 12)      { HH[l] = 1; KK[l] = 1; }
    else if (l == 12){ HH[l] = 2; KK[l] = 1; }
    else if (l < 24) { HH[l] = 2; KK[l] = 2; }
    else if (l == 24){ HH[l] = 4; KK[l] = 2; }
    else             { HH[l] = 4; KK[l] = 4; }
  }
  pack_conv1_kernel<<<1, 512, 0, stream>>>(conv1w, pack);
  size_t packoff[36];
  size_t poff = 512;
  int offz = 0;
  for (int l = 0; l < 36; ++l) {
    int K  = KK[l] * 16 * 9;
    int KT = (K + 31) / 32;
    int elems = HH[l] * KT * 512;
    packoff[l] = poff;
    pack_layer_kernel<<<(elems + 255) / 256, 256, 0, stream>>>(kflat, pack + poff,
                                                               offz, HH[l], KK[l]);
    poff += (size_t)elems;
    offz += HH[l] * KK[l];
  }

  // ---- input convert + conv1 + bn0 --------------------------------------
  cvt_x_kernel<<<(1990656 + 255) / 256, 256, 0, stream>>>(x, xbf, 1990656LL);

  auto launch_conv = [&](int kind, const unsigned short* a, const unsigned short* wp, float* o) {
    switch (kind) {
      case 0: conv_wmma_kernel<3, 16, 36, 36, 1, 36, 36><<<663552 / 128, 256, 0, stream>>>(a, wp, o); break;
      case 1: conv_wmma_kernel<16, 16, 36, 36, 1, 36, 36><<<663552 / 128, 256, 0, stream>>>(a, wp, o); break;
      case 2: conv_wmma_kernel<16, 32, 36, 36, 2, 18, 18><<<165888 / 128, 256, 0, stream>>>(a, wp, o); break;
      case 3: conv_wmma_kernel<32, 32, 18, 18, 1, 18, 18><<<165888 / 128, 256, 0, stream>>>(a, wp, o); break;
      case 4: conv_wmma_kernel<32, 64, 18, 18, 2, 9, 9><<<41472 / 128, 256, 0, stream>>>(a, wp, o); break;
      case 5: conv_wmma_kernel<64, 64, 9, 9, 1, 9, 9><<<41472 / 128, 256, 0, stream>>>(a, wp, o); break;
    }
  };
  auto run_bn = [&](float* conv, int C, int HW, int bnj, const unsigned short* resid,
                    unsigned short* outa, long long total) {
    bn_reduce_kernel<<<BN_SPLIT, 256, 0, stream>>>(conv, part, C, total);
    bn_finalize_kernel<<<1, 64, 0, stream>>>(part, gam[bnj], bet[bnj], bnsc, bnsh, C, HW);
    bn_apply_kernel<<<2048, 256, 0, stream>>>(conv, resid, bnsc, bnsh, outa, total, C - 1);
  };

  launch_conv(0, xbf, pack, convb);
  run_bn(convb, 16, 1296, 0, nullptr, act0, 10616832LL);

  // ---- 18 residual blocks ------------------------------------------------
  for (int i = 0; i < 18; ++i) {
    const bool down = (i == 6 || i == 12);
    const int Cout  = (i < 6) ? 16 : ((i < 12) ? 32 : 64);
    const int HOUT  = (i < 6) ? 36 : ((i < 12) ? 18 : 9);
    const int HWo   = HOUT * HOUT;
    const long long total = 512LL * Cout * HWo;
    const int kindA = (i < 6) ? 1 : (i == 6 ? 2 : (i < 12 ? 3 : (i == 12 ? 4 : 5)));
    const int kindB = (i < 6) ? 1 : ((i < 12) ? 3 : 5);

    if (i == 6)
      res_conv_kernel<<<(unsigned)((5308416LL + 255) / 256), 256, 0, stream>>>(
          act0, res1w, res1b, resb, 16, 36, 36, 32, 18, 18, 5308416LL);
    if (i == 12)
      res_conv_kernel<<<(unsigned)((2654208LL + 255) / 256), 256, 0, stream>>>(
          act0, res2w, res2b, resb, 32, 18, 18, 64, 9, 9, 2654208LL);

    launch_conv(kindA, act0, pack + packoff[2 * i], convb);
    run_bn(convb, Cout, HWo, 1 + 2 * i, nullptr, act1, total);

    launch_conv(kindB, act1, pack + packoff[2 * i + 1], convb);
    run_bn(convb, Cout, HWo, 2 + 2 * i, down ? resb : act0, act0, total);
  }

  // ---- global average pool + linear head ---------------------------------
  final_head_kernel<<<512, 64, 0, stream>>>(act0, finW, finB, (float*)d_out);
}
